// modi_cgcnn_a2e_46248207843557
// MI455X (gfx1250) — matrix operand order repaired and checked by reading, not compiled, not run on hardware
//
#include <hip/hip_runtime.h>
#include <hip/hip_bf16.h>
#include <math.h>

constexpr int FD = 128;        // F
constexpr int AD = 64;         // A
constexpr int ND = 256;        // 2F  (GEMM N)
constexpr int KD = 320;        // 2F+A (GEMM K)
constexpr int KSTEPS = KD / 32;   // 10
constexpr int NTILES = ND / 16;   // 16
constexpr int N_EDGES  = 100000;
constexpr int N_ANGLES = 400000;
constexpr int N_CRYST  = 512;
constexpr float EPSV = 1e-5f;
constexpr float INV_SQRT2 = 0.70710678118654752440f;

typedef __attribute__((ext_vector_type(16))) __bf16 v16bf;
typedef __attribute__((ext_vector_type(16))) unsigned short v16u;
typedef __attribute__((ext_vector_type(8)))  float v8f;

__device__ __forceinline__ unsigned short f32_bf16(float f) {
    unsigned int u = __float_as_uint(f);
    u += 0x7fffu + ((u >> 16) & 1u);      // round-to-nearest-even
    return (unsigned short)(u >> 16);
}

// ---------------- zero workspace region ----------------
__global__ void zero_kernel(float* __restrict__ p, size_t n) {
    size_t i = (size_t)blockIdx.x * blockDim.x + threadIdx.x;
    size_t st = (size_t)gridDim.x * blockDim.x;
    for (; i < n; i += st) p[i] = 0.0f;
}

// ---------------- pack W_full into WMMA B-fragment layout ----------------
// Fragment (nt, ks): lane l holds B[k][n] = W[n][k] for n = nt*16 + (l&15),
// k = ks*32 + 16*(l>>4) + j,  j = 0..15 (contiguous K per lane).
__global__ void pack_w_kernel(const float* __restrict__ Wfull,
                              unsigned short* __restrict__ Bpack) {
    int idx = blockIdx.x * blockDim.x + threadIdx.x;
    if (idx >= ND * KD) return;
    int j    = idx & 15;
    int lane = (idx >> 4) & 31;
    int ks   = (idx >> 9) % KSTEPS;
    int nt   = idx / (512 * KSTEPS);
    int n = nt * 16 + (lane & 15);
    int k = ks * 32 + ((lane >> 4) << 4) + j;
    Bpack[idx] = f32_bf16(Wfull[(size_t)n * KD + k]);
}

// ---------------- main GEMM: h[400k,256] = gathered X[400k,320] @ W^T ----------------
__global__ __launch_bounds__(256) void gemm_kernel(
    const float* __restrict__ angle_fea,
    const float* __restrict__ nbr_fea,
    const int*   __restrict__ nbr_idx,
    const unsigned short* __restrict__ Bpack,
    float* __restrict__ h)
{
    const int lane  = threadIdx.x & 31;
    const int wave  = threadIdx.x >> 5;
    const int mtile = blockIdx.x * 8 + wave;
    const int m0 = mtile << 4;
    const int g  = lane >> 4;      // lane group
    const int mr = lane & 15;
    const int m  = m0 + mr;

    const int i0 = nbr_idx[2 * m];
    const int i1 = nbr_idx[2 * m + 1];
    const float* __restrict__ arow = angle_fea + (size_t)m  * AD;
    const float* __restrict__ n0p  = nbr_fea   + (size_t)i0 * FD;
    const float* __restrict__ n1p  = nbr_fea   + (size_t)i1 * FD;

    // A fragments: contiguous K per lane half, same order as Bpack:
    // elements (b = half*8 + j) hold K = ks*32 + g*16 + half*8 + j
    v16u afrag[KSTEPS];
#pragma unroll
    for (int ks = 0; ks < KSTEPS; ++ks) {
#pragma unroll
        for (int half = 0; half < 2; ++half) {
            int k = ks * 32 + g * 16 + half * 8;   // multiple of 8, never straddles 64/192
            const float* p;
            if (k < 64)       p = arow + k;
            else if (k < 192) p = n0p + (k - 64);
            else              p = n1p + (k - 192);
            float4 lo = *(const float4*)p;
            float4 hi = *(const float4*)(p + 4);
            int b = half * 8;
            afrag[ks][b + 0] = f32_bf16(lo.x); afrag[ks][b + 1] = f32_bf16(lo.y);
            afrag[ks][b + 2] = f32_bf16(lo.z); afrag[ks][b + 3] = f32_bf16(lo.w);
            afrag[ks][b + 4] = f32_bf16(hi.x); afrag[ks][b + 5] = f32_bf16(hi.y);
            afrag[ks][b + 6] = f32_bf16(hi.z); afrag[ks][b + 7] = f32_bf16(hi.w);
        }
    }

    const v16u* __restrict__ Bf = (const v16u*)Bpack;
#pragma unroll 1
    for (int nt = 0; nt < NTILES; ++nt) {
        v8f c = {0.f, 0.f, 0.f, 0.f, 0.f, 0.f, 0.f, 0.f};
#pragma unroll
        for (int ks = 0; ks < KSTEPS; ++ks) {
            v16u bf = Bf[(nt * KSTEPS + ks) * 32 + lane];
            c = __builtin_amdgcn_wmma_f32_16x16x32_bf16(
                    false, __builtin_bit_cast(v16bf, afrag[ks]),
                    false, __builtin_bit_cast(v16bf, bf),
                    (short)0, c, false, false);
        }
        // C layout: element r -> (M = r + 8*g, N = lane&15)
        float* hp = h + (size_t)(m0 + g * 8) * ND + nt * 16 + mr;
#pragma unroll
        for (int r = 0; r < 8; ++r) hp[(size_t)r * ND] = c[r];
    }
}

// ---------------- segment stats (crystal_angle_index is sorted) ----------------
__global__ __launch_bounds__(256) void seg_stats_kernel(
    const float* __restrict__ h, const int* __restrict__ seg,
    float* __restrict__ ssum, float* __restrict__ ssq, float* __restrict__ scnt)
{
    const int t    = threadIdx.x;
    const int base = blockIdx.x * 256;
    const int end  = (base + 256 < N_ANGLES) ? base + 256 : N_ANGLES;
    float aS = 0.f, aQ = 0.f; int aC = 0;
    int cur = -1;
    for (int row = base; row < end; ++row) {
        int sg = seg[row];
        if (sg != cur) {
            if (cur >= 0) {
                atomicAdd(&ssum[(size_t)cur * ND + t], aS);
                atomicAdd(&ssq [(size_t)cur * ND + t], aQ);
                if (t == 0) atomicAdd(&scnt[cur], (float)aC);
            }
            cur = sg; aS = 0.f; aQ = 0.f; aC = 0;
        }
        float v = h[(size_t)row * ND + t];
        aS += v; aQ += v * v; ++aC;
    }
    if (cur >= 0) {
        atomicAdd(&ssum[(size_t)cur * ND + t], aS);
        atomicAdd(&ssq [(size_t)cur * ND + t], aQ);
        if (t == 0) atomicAdd(&scnt[cur], (float)aC);
    }
}

__global__ void seg_final_kernel(const float* __restrict__ ssum,
                                 const float* __restrict__ ssq,
                                 const float* __restrict__ scnt,
                                 float* __restrict__ meanb, float* __restrict__ invb)
{
    int idx = blockIdx.x * blockDim.x + threadIdx.x;
    if (idx >= N_CRYST * ND) return;
    int s = idx / ND;
    float c  = fmaxf(scnt[s], 1.0f);
    float mu = ssum[idx] / c;
    float var = fmaxf(ssq[idx] / c - mu * mu, 0.0f);
    meanb[idx] = mu;
    invb[idx]  = 1.0f / sqrtf(var + EPSV);
}

// ---------------- per-angle: apply norm, LN+SiLU, gate, scatter ----------------
__global__ __launch_bounds__(128) void angle_msg_kernel(
    const float* __restrict__ h, const int* __restrict__ seg,
    const int* __restrict__ nbr_idx,
    const float* __restrict__ meanb, const float* __restrict__ invb,
    const float* __restrict__ cng, const float* __restrict__ cnb,
    const float* __restrict__ lng, const float* __restrict__ lnb,
    const float* __restrict__ wmask,
    float* __restrict__ s_out, float* __restrict__ c_out)
{
    const int row = blockIdx.x;
    const int t   = threadIdx.x;
    const int sg  = seg[row];
    const size_t hb = (size_t)row * ND;
    const size_t mi = (size_t)sg * ND;
    float hc = h[hb + t];
    float hf = h[hb + FD + t];
    float core = (hc - meanb[mi + t])      * invb[mi + t]      * cng[t]      + cnb[t];
    float filt = (hf - meanb[mi + FD + t]) * invb[mi + FD + t] * cng[FD + t] + cnb[FD + t];

    __shared__ float r0[128], r1[128], r2[128];
    r0[t] = core; r1[t] = core * core; r2[t] = filt * wmask[t];
    __syncthreads();
    for (int st = 64; st > 0; st >>= 1) {
        if (t < st) { r0[t] += r0[t + st]; r1[t] += r1[t + st]; r2[t] += r2[t + st]; }
        __syncthreads();
    }
    float mu  = r0[0] * (1.0f / FD);
    float var = fmaxf(r1[0] * (1.0f / FD) - mu * mu, 0.0f);
    float x   = (core - mu) / sqrtf(var + EPSV) * lng[t] + lnb[t];
    float sil = x / (1.0f + expf(-x));
    float gate = 1.0f / (1.0f + expf(-r2[0]));
    float msg = gate * sil;

    int src = nbr_idx[2 * row];
    atomicAdd(&s_out[(size_t)src * FD + t], msg);
    if (t == 0) atomicAdd(&c_out[src], 1.0f);
}

// ---------------- per-edge: mean, LN, two residual MLPs, output ----------------
__global__ __launch_bounds__(128) void edge_final_kernel(
    const float* __restrict__ s_in, const float* __restrict__ c_in,
    const float* __restrict__ nbr_fea,
    const float* __restrict__ g2, const float* __restrict__ b2,
    const float* __restrict__ W1a, const float* __restrict__ b1a,
    const float* __restrict__ W2a, const float* __restrict__ b2a,
    const float* __restrict__ W1b, const float* __restrict__ b1b,
    const float* __restrict__ W2b, const float* __restrict__ b2b,
    float* __restrict__ out)
{
    const int e = blockIdx.x;
    const int t = threadIdx.x;
    float cnt = fmaxf(c_in[e], 1.0f);
    float v = s_in[(size_t)e * FD + t] / cnt;

    __shared__ float r0[128], r1[128];
    r0[t] = v; r1[t] = v * v;
    __syncthreads();
    for (int st = 64; st > 0; st >>= 1) {
        if (t < st) { r0[t] += r0[t + st]; r1[t] += r1[t + st]; }
        __syncthreads();
    }
    float mu  = r0[0] * (1.0f / FD);
    float var = fmaxf(r1[0] * (1.0f / FD) - mu * mu, 0.0f);
    float x = (v - mu) / sqrtf(var + EPSV) * g2[t] + b2[t];

    __shared__ float xs[128];
    __shared__ float hid[64];

    // residual block a
    xs[t] = x; __syncthreads();
    if (t < 64) {
        float acc = b1a[t];
        const float* w = W1a + (size_t)t * FD;
#pragma unroll 8
        for (int f = 0; f < FD; ++f) acc += xs[f] * w[f];
        hid[t] = acc / (1.0f + expf(-acc));
    }
    __syncthreads();
    {
        float acc = b2a[t];
        const float* w = W2a + (size_t)t * 64;
#pragma unroll 8
        for (int j = 0; j < 64; ++j) acc += hid[j] * w[j];
        x += acc;
    }
    __syncthreads();

    // residual block b
    xs[t] = x; __syncthreads();
    if (t < 64) {
        float acc = b1b[t];
        const float* w = W1b + (size_t)t * FD;
#pragma unroll 8
        for (int f = 0; f < FD; ++f) acc += xs[f] * w[f];
        hid[t] = acc / (1.0f + expf(-acc));
    }
    __syncthreads();
    {
        float acc = b2b[t];
        const float* w = W2b + (size_t)t * 64;
#pragma unroll 8
        for (int j = 0; j < 64; ++j) acc += hid[j] * w[j];
        x += acc;
    }

    out[(size_t)e * FD + t] = (nbr_fea[(size_t)e * FD + t] + x) * INV_SQRT2;
}

// ---------------- launch ----------------
extern "C" void kernel_launch(void* const* d_in, const int* in_sizes, int n_in,
                              void* d_out, int out_size, void* d_ws, size_t ws_size,
                              hipStream_t stream)
{
    const float* nbr_fea   = (const float*)d_in[0];
    const float* angle_fea = (const float*)d_in[1];
    const int*   nbr_idx   = (const int*)  d_in[2];
    /* d_in[3] crystal_edge_idx: unused by the reference */
    const int*   seg       = (const int*)  d_in[4];
    const float* W_full    = (const float*)d_in[5];
    const float* W_mask    = (const float*)d_in[6];
    const float* cn_g      = (const float*)d_in[7];
    const float* cn_b      = (const float*)d_in[8];
    const float* lnc_g     = (const float*)d_in[9];
    const float* lnc_b     = (const float*)d_in[10];
    const float* ln2_g     = (const float*)d_in[11];
    const float* ln2_b     = (const float*)d_in[12];
    const float* W1a = (const float*)d_in[13];
    const float* b1a = (const float*)d_in[14];
    const float* W2a = (const float*)d_in[15];
    const float* b2a = (const float*)d_in[16];
    const float* W1b = (const float*)d_in[17];
    const float* b1b = (const float*)d_in[18];
    const float* W2b = (const float*)d_in[19];
    const float* b2b = (const float*)d_in[20];

    char* ws = (char*)d_ws;
    size_t off = 0;
    auto wsalloc = [&](size_t bytes) -> void* {
        void* p = ws + off;
        off += (bytes + 255) & ~(size_t)255;
        return p;
    };
    unsigned short* Bpack = (unsigned short*)wsalloc((size_t)ND * KD * 2);
    float* h     = (float*)wsalloc((size_t)N_ANGLES * ND * 4);
    float* ssum  = (float*)wsalloc((size_t)N_CRYST * ND * 4);   // zeroed region start
    float* ssq   = (float*)wsalloc((size_t)N_CRYST * ND * 4);
    float* scnt  = (float*)wsalloc((size_t)N_CRYST * 4);
    float* sacc  = (float*)wsalloc((size_t)N_EDGES * FD * 4);
    float* cacc  = (float*)wsalloc((size_t)N_EDGES * 4);        // zeroed region end
    float* meanb = (float*)wsalloc((size_t)N_CRYST * ND * 4);
    float* invb  = (float*)wsalloc((size_t)N_CRYST * ND * 4);

    size_t zero_floats = (size_t)((char*)meanb - (char*)ssum) / 4;
    zero_kernel<<<2048, 256, 0, stream>>>(ssum, zero_floats);

    pack_w_kernel<<<(ND * KD + 255) / 256, 256, 0, stream>>>(W_full, Bpack);

    gemm_kernel<<<(N_ANGLES / 16) / 8, 256, 0, stream>>>(angle_fea, nbr_fea, nbr_idx, Bpack, h);

    seg_stats_kernel<<<(N_ANGLES + 255) / 256, 256, 0, stream>>>(h, seg, ssum, ssq, scnt);

    seg_final_kernel<<<(N_CRYST * ND + 255) / 256, 256, 0, stream>>>(ssum, ssq, scnt, meanb, invb);

    angle_msg_kernel<<<N_ANGLES, 128, 0, stream>>>(h, seg, nbr_idx, meanb, invb,
                                                   cn_g, cn_b, lnc_g, lnc_b, W_mask,
                                                   sacc, cacc);

    edge_final_kernel<<<N_EDGES, 128, 0, stream>>>(sacc, cacc, nbr_fea, ln2_g, ln2_b,
                                                   W1a, b1a, W2a, b2a,
                                                   W1b, b1b, W2b, b2b,
                                                   (float*)d_out);
}
